// TransformerEncoderLayer_18829136626004
// MI455X (gfx1250) — matrix operand order repaired and checked by reading, not compiled, not run on hardware
//
#include <hip/hip_runtime.h>
#include <hip/hip_bf16.h>

// ---------------------------------------------------------------------------
// Types & constants
// ---------------------------------------------------------------------------
typedef __bf16 bf16;
typedef bf16  v16bf  __attribute__((ext_vector_type(16)));
typedef bf16  bf16x2 __attribute__((ext_vector_type(2)));
typedef float v8f    __attribute__((ext_vector_type(8)));

constexpr int  Sd   = 2048;   // sequence length
constexpr int  Dd   = 1024;   // model dim
constexpr int  Hd   = 16;     // heads
constexpr int  DHd  = 64;     // head dim
constexpr int  DFFd = 4096;   // ffn dim
constexpr int  Md   = 2 * Sd; // B*S = 4096 rows

// ---------------------------------------------------------------------------
// bf16 convert helpers — native casts so the backend can use v_cvt_*bf16*
// ---------------------------------------------------------------------------
__device__ __forceinline__ bf16  f2bf(float f) { return (bf16)f; }
__device__ __forceinline__ float bf2f(bf16 b)  { return (float)b; }
__device__ __forceinline__ unsigned pack2(float a, float b) {
  bf16x2 t = { (bf16)a, (bf16)b };
  return __builtin_bit_cast(unsigned, t);
}
// 4 contiguous source elems -> 4 packed bf16 (8 bytes)
__device__ __forceinline__ uint2 cvt4(const float* p) {
  float4 v = *(const float4*)p;
  return make_uint2(pack2(v.x, v.y), pack2(v.z, v.w));
}
__device__ __forceinline__ uint2 cvt4(const bf16* p) {
  return *(const uint2*)p;   // already bf16: raw 8-byte copy
}

// ---------------------------------------------------------------------------
// WMMA fragment builders.
// A-fragment (16x32 bf16): lane(row=lane%16, half=lane/16) holds
//   k = 16*half + e (contiguous 32B)
// B-fragment (32x16 bf16): lane(col=lane%16, half) holds k = 16*half + e
//   (contiguous 32B). Both built from k-contiguous rows.
// ---------------------------------------------------------------------------
union FragU { v16bf v; uint2 u2[4]; };

__device__ __forceinline__ v16bf lds_frag_a(const bf16* rowp, int half) {
  FragU t;
  t.u2[0] = *(const uint2*)(rowp + 16 * half);
  t.u2[1] = *(const uint2*)(rowp + 16 * half + 4);
  t.u2[2] = *(const uint2*)(rowp + 16 * half + 8);
  t.u2[3] = *(const uint2*)(rowp + 16 * half + 12);
  return t.v;
}
__device__ __forceinline__ v16bf lds_frag_b(const bf16* rowp, int half) {
  const uint2* p = (const uint2*)(rowp + 16 * half);
  FragU t;
  t.u2[0] = p[0]; t.u2[1] = p[1]; t.u2[2] = p[2]; t.u2[3] = p[3];
  return t.v;
}
// from global f32, 16 contiguous elements (B-fragment pattern)
__device__ __forceinline__ v16bf g_frag16(const float* p) {
  FragU t;
  t.u2[0] = cvt4(p); t.u2[1] = cvt4(p + 4);
  t.u2[2] = cvt4(p + 8); t.u2[3] = cvt4(p + 12);
  return t.v;
}
// from global f32, A-fragment pattern over a 32-wide row
__device__ __forceinline__ v16bf g_frag_a(const float* p, int half) {
  FragU t;
  t.u2[0] = cvt4(p + 16 * half);     t.u2[1] = cvt4(p + 16 * half + 4);
  t.u2[2] = cvt4(p + 16 * half + 8); t.u2[3] = cvt4(p + 16 * half + 12);
  return t.v;
}

__device__ __forceinline__ v8f wmma_bf16(v16bf a, v16bf b, v8f c) {
  return __builtin_amdgcn_wmma_f32_16x16x32_bf16(false, a, false, b,
                                                 (short)0, c, false, false);
}

// ---------------------------------------------------------------------------
// Conv1d (width 3, same padding) as WMMA GEMM, fused bias + ReLU.
// ---------------------------------------------------------------------------
__global__ __launch_bounds__(128)
void conv_gemm_kernel(const float* __restrict__ X, const float* __restrict__ W,
                      const float* __restrict__ bias, float* __restrict__ Y) {
  __shared__ bf16 As[64][32];    // rows x k (k contiguous)
  __shared__ bf16 Bst[64][32];   // cols x k (k contiguous)
  const int tid  = threadIdx.x;
  const int wave = tid >> 5, lane = tid & 31;
  const int half = lane >> 4, l16 = lane & 15;
  const int row0 = blockIdx.x * 64;
  const int col0 = blockIdx.y * 64;

  v8f acc[4] = {};
  for (int w = 0; w < 3; ++w) {
    for (int k0 = 0; k0 < Dd; k0 += 32) {
      __syncthreads();
      // Stage A: 512 float4 chunks, 4 per thread; shift + zero pad per row
      #pragma unroll
      for (int it = 0; it < 4; ++it) {
        int idx = tid + it * 128;          // 0..511
        int r = idx >> 3, c4 = (idx & 7) << 2;
        int gr = row0 + r;
        int b = gr >> 11;
        int s = (gr & (Sd - 1)) + w - 1;
        uint2 val = make_uint2(0u, 0u);
        if (s >= 0 && s < Sd)
          val = cvt4(&X[((size_t)(b << 11) + s) * Dd + k0 + c4]);
        *(uint2*)&As[r][c4] = val;
      }
      // Stage B transposed: W[o][i][w] at o*D*3 + i*3 + w (stride-3 gather)
      #pragma unroll
      for (int it = 0; it < 4; ++it) {
        int idx = tid + it * 128;
        int n = idx >> 3, k4 = (idx & 7) << 2;
        size_t base = ((size_t)(col0 + n) * Dd + (k0 + k4)) * 3 + w;
        float a0 = W[base], a1 = W[base + 3], a2 = W[base + 6], a3 = W[base + 9];
        *(uint2*)&Bst[n][k4] = make_uint2(pack2(a0, a1), pack2(a2, a3));
      }
      __syncthreads();

      v16bf af = lds_frag_a(&As[wave * 16 + l16][0], half);
      #pragma unroll
      for (int nt = 0; nt < 4; ++nt) {
        v16bf bfv = lds_frag_b(&Bst[nt * 16 + l16][0], half);
        acc[nt] = wmma_bf16(af, bfv, acc[nt]);
      }
    }
  }
  // Epilogue: bias + ReLU. C layout: row = r + 8*half, col = lane%16
  #pragma unroll
  for (int nt = 0; nt < 4; ++nt) {
    #pragma unroll
    for (int r = 0; r < 8; ++r) {
      int grow = row0 + wave * 16 + r + half * 8;
      int gcol = col0 + nt * 16 + l16;
      float v = acc[nt][r] + bias[gcol];
      v = v > 0.0f ? v : 0.0f;
      Y[(size_t)grow * Dd + gcol] = v;
    }
  }
}

// ---------------------------------------------------------------------------
// Generic WMMA GEMM: Y = f(alpha * A @ W^T + bias) * mask + resid
// A: M x K (type AT), W: N x K (f32, row-major), Y: M x N (type OT)
// ---------------------------------------------------------------------------
template <typename AT, typename OT>
__global__ __launch_bounds__(128)
void gemm_wmma_kernel(const AT* __restrict__ A, const float* __restrict__ W,
                      OT* __restrict__ Y, int N, int K, float alpha,
                      const float* __restrict__ bias, int relu,
                      const float* __restrict__ mask,
                      const float* __restrict__ resid) {
  __shared__ bf16 As[64][32];
  __shared__ bf16 Bst[64][32];
  const int tid  = threadIdx.x;
  const int wave = tid >> 5, lane = tid & 31;
  const int half = lane >> 4, l16 = lane & 15;
  const int row0 = blockIdx.x * 64;
  const int col0 = blockIdx.y * 64;

  v8f acc[4] = {};
  for (int k0 = 0; k0 < K; k0 += 32) {
    __syncthreads();
    #pragma unroll
    for (int it = 0; it < 4; ++it) {
      int idx = tid + it * 128;
      int r = idx >> 3, c4 = (idx & 7) << 2;
      *(uint2*)&As[r][c4] = cvt4(&A[(size_t)(row0 + r) * K + k0 + c4]);
    }
    #pragma unroll
    for (int it = 0; it < 4; ++it) {
      int idx = tid + it * 128;
      int n = idx >> 3, k4 = (idx & 7) << 2;
      *(uint2*)&Bst[n][k4] = cvt4(&W[(size_t)(col0 + n) * K + k0 + k4]);
    }
    __syncthreads();

    v16bf af = lds_frag_a(&As[wave * 16 + l16][0], half);
    #pragma unroll
    for (int nt = 0; nt < 4; ++nt) {
      v16bf bfv = lds_frag_b(&Bst[nt * 16 + l16][0], half);
      acc[nt] = wmma_bf16(af, bfv, acc[nt]);
    }
  }
  #pragma unroll
  for (int nt = 0; nt < 4; ++nt) {
    #pragma unroll
    for (int r = 0; r < 8; ++r) {
      int grow = row0 + wave * 16 + r + half * 8;
      int gcol = col0 + nt * 16 + l16;
      float v = acc[nt][r] * alpha;
      if (bias)  v += bias[gcol];
      if (relu)  v = v > 0.0f ? v : 0.0f;
      if (mask)  v *= mask[grow];
      if (resid) v += resid[(size_t)grow * N + gcol];
      if constexpr (__is_same(OT, bf16)) Y[(size_t)grow * N + gcol] = f2bf(v);
      else                               Y[(size_t)grow * N + gcol] = v;
    }
  }
}

// ---------------------------------------------------------------------------
// LayerNorm over D=1024, one row per 256-thread block, float4 I/O
// ---------------------------------------------------------------------------
__device__ __forceinline__ float block_sum(float v, float* red) {
  int tid = threadIdx.x;
  red[tid] = v;
  __syncthreads();
  for (int s = 128; s > 0; s >>= 1) {
    if (tid < s) red[tid] += red[tid + s];
    __syncthreads();
  }
  float r = red[0];
  __syncthreads();
  return r;
}

__global__ __launch_bounds__(256)
void ln_kernel(const float* __restrict__ X, float* __restrict__ Y,
               const float* __restrict__ g, const float* __restrict__ b) {
  __shared__ float red[256];
  size_t row = blockIdx.x;
  int c = threadIdx.x * 4;
  float4 v = *(const float4*)(X + row * Dd + c);
  float mu = block_sum(v.x + v.y + v.z + v.w, red) * (1.0f / Dd);
  float dx = v.x - mu, dy = v.y - mu, dz = v.z - mu, dw = v.w - mu;
  float var = block_sum(dx * dx + dy * dy + dz * dz + dw * dw, red) * (1.0f / Dd);
  float inv = rsqrtf(var + 1e-6f);
  float4 gg = *(const float4*)(g + c);
  float4 bb = *(const float4*)(b + c);
  float4 o;
  o.x = dx * inv * gg.x + bb.x; o.y = dy * inv * gg.y + bb.y;
  o.z = dz * inv * gg.z + bb.z; o.w = dw * inv * gg.w + bb.w;
  *(float4*)(Y + row * Dd + c) = o;
}

// inputs = Hc + SC ; XN = LayerNorm(inputs)
__global__ __launch_bounds__(256)
void add_ln_kernel(const float* __restrict__ Hc, const float* __restrict__ SC,
                   float* __restrict__ INP, float* __restrict__ XN,
                   const float* __restrict__ g, const float* __restrict__ b) {
  __shared__ float red[256];
  size_t row = blockIdx.x;
  int c = threadIdx.x * 4;
  float4 a = *(const float4*)(Hc + row * Dd + c);
  float4 s = *(const float4*)(SC + row * Dd + c);
  float4 v;
  v.x = a.x + s.x; v.y = a.y + s.y; v.z = a.z + s.z; v.w = a.w + s.w;
  *(float4*)(INP + row * Dd + c) = v;
  float mu = block_sum(v.x + v.y + v.z + v.w, red) * (1.0f / Dd);
  float dx = v.x - mu, dy = v.y - mu, dz = v.z - mu, dw = v.w - mu;
  float var = block_sum(dx * dx + dy * dy + dz * dz + dw * dw, red) * (1.0f / Dd);
  float inv = rsqrtf(var + 1e-6f);
  float4 gg = *(const float4*)(g + c);
  float4 bb = *(const float4*)(b + c);
  float4 o;
  o.x = dx * inv * gg.x + bb.x; o.y = dy * inv * gg.y + bb.y;
  o.z = dz * inv * gg.z + bb.z; o.w = dw * inv * gg.w + bb.w;
  *(float4*)(XN + row * Dd + c) = o;
}

// ---------------------------------------------------------------------------
// Flash attention: one wave per (b, h, 16-query tile). Q pre-scaled by 1/8.
// Scores via WMMA (K fragments = contiguous global b128 loads); online
// softmax with 16-lane shuffles; V staged transposed in LDS; ctx via WMMA.
// ---------------------------------------------------------------------------
__global__ __launch_bounds__(32)
void attn_kernel(const float* __restrict__ Q, const float* __restrict__ Km,
                 const float* __restrict__ V, const unsigned char* __restrict__ msk,
                 float* __restrict__ O) {
  const int lane = threadIdx.x & 31;
  const int half = lane >> 4, l16 = lane & 15;
  const int qt = blockIdx.x & 127;            // S/16 = 128
  const int h  = (blockIdx.x >> 7) & 15;
  const int b  = blockIdx.x >> 11;
  const size_t rowbase = (size_t)b * Sd;
  const int hoff = h * DHd;

  // Q A-fragments (16 queries x 64 dh = 2 K-chunks of 32)
  v16bf qf[2];
  const int qrow = qt * 16 + l16;
  #pragma unroll
  for (int c = 0; c < 2; ++c)
    qf[c] = g_frag_a(&Q[(rowbase + qrow) * Dd + hoff + c * 32], half);

  float m_run[8], l_run[8];
  #pragma unroll
  for (int r = 0; r < 8; ++r) { m_run[r] = -1e30f; l_run[r] = 0.0f; }
  v8f ctx[4] = {};

  __shared__ bf16 Pl[16][32];   // probs: row x key (A-fragment pattern)
  __shared__ bf16 Vs[64][32];   // V transposed: dim x key (B-fragment pattern)

  for (int kt = 0; kt < Sd / 32; ++kt) {
    // ---- scores: two 16x16 tiles over keys kt*32+[0..15], +[16..31] ----
    v8f s0 = {}, s1 = {};
    #pragma unroll
    for (int c = 0; c < 2; ++c) {
      v16bf kf0 = g_frag16(&Km[(rowbase + kt * 32 + l16)      * Dd + hoff + c * 32 + 16 * half]);
      v16bf kf1 = g_frag16(&Km[(rowbase + kt * 32 + 16 + l16) * Dd + hoff + c * 32 + 16 * half]);
      s0 = wmma_bf16(qf[c], kf0, s0);
      s1 = wmma_bf16(qf[c], kf1, s1);
    }
    // ---- stage V tile transposed: 512 float4 chunks, 16 per lane ----
    #pragma unroll
    for (int it = 0; it < 16; ++it) {
      int idx = lane + it * 32;         // 0..511
      int key = idx >> 4, d4 = (idx & 15) << 2;
      float4 vv = *(const float4*)&V[(rowbase + kt * 32 + key) * Dd + hoff + d4];
      Vs[d4 + 0][key] = f2bf(vv.x);
      Vs[d4 + 1][key] = f2bf(vv.y);
      Vs[d4 + 2][key] = f2bf(vv.z);
      Vs[d4 + 3][key] = f2bf(vv.w);
    }
    // ---- mask (attn_mask is (B,1,S) bool) ----
    bool k0v = msk[(size_t)b * Sd + kt * 32 + l16]      != 0;
    bool k1v = msk[(size_t)b * Sd + kt * 32 + 16 + l16] != 0;
    #pragma unroll
    for (int r = 0; r < 8; ++r) {
      s0[r] = k0v ? s0[r] : -1e9f;
      s1[r] = k1v ? s1[r] : -1e9f;
    }
    // ---- online softmax + stage P (C layout -> A layout via LDS) ----
    #pragma unroll
    for (int r = 0; r < 8; ++r) {
      float mx = fmaxf(s0[r], s1[r]);
      #pragma unroll
      for (int off = 8; off >= 1; off >>= 1) mx = fmaxf(mx, __shfl_xor(mx, off, 16));
      float nm = fmaxf(m_run[r], mx);
      float sc = __expf(m_run[r] - nm);
      m_run[r] = nm;
      float p0 = __expf(s0[r] - nm);
      float p1 = __expf(s1[r] - nm);
      float ps = p0 + p1;
      #pragma unroll
      for (int off = 8; off >= 1; off >>= 1) ps += __shfl_xor(ps, off, 16);
      l_run[r] = l_run[r] * sc + ps;
      #pragma unroll
      for (int nt = 0; nt < 4; ++nt) ctx[nt][r] *= sc;
      Pl[r + half * 8][l16]      = f2bf(p0);
      Pl[r + half * 8][16 + l16] = f2bf(p1);
    }
    __syncthreads();  // single-wave block: orders LDS writes vs fragment reads
    // ---- ctx += P(16x32) @ V(32x64) ----
    v16bf pf = lds_frag_a(&Pl[l16][0], half);
    #pragma unroll
    for (int nt = 0; nt < 4; ++nt) {
      v16bf vf = lds_frag_b(&Vs[nt * 16 + l16][0], half);
      ctx[nt] = wmma_bf16(pf, vf, ctx[nt]);
    }
    __syncthreads();
  }
  // ---- finalize: divide by l, scatter to (B*S, D) ----
  #pragma unroll
  for (int nt = 0; nt < 4; ++nt)
    #pragma unroll
    for (int r = 0; r < 8; ++r) {
      int s = qt * 16 + r + half * 8;
      O[(rowbase + s) * Dd + hoff + nt * 16 + l16] = ctx[nt][r] / l_run[r];
    }
}

// ---------------------------------------------------------------------------
// Launch
// ---------------------------------------------------------------------------
extern "C" void kernel_launch(void* const* d_in, const int* in_sizes, int n_in,
                              void* d_out, int out_size, void* d_ws, size_t ws_size,
                              hipStream_t stream) {
  const float* x     = (const float*)d_in[0];
  const unsigned char* amask = (const unsigned char*)d_in[1];
  const float* fmask = (const float*)d_in[2];
  const float* c1w   = (const float*)d_in[3];
  const float* c1b   = (const float*)d_in[4];
  const float* ln1g  = (const float*)d_in[5];
  const float* ln1b  = (const float*)d_in[6];
  const float* c2w   = (const float*)d_in[7];
  const float* c2b   = (const float*)d_in[8];
  const float* ln2g  = (const float*)d_in[9];
  const float* ln2b  = (const float*)d_in[10];
  const float* lnAg  = (const float*)d_in[11];
  const float* lnAb  = (const float*)d_in[12];
  const float* Wq    = (const float*)d_in[13];
  const float* Wk    = (const float*)d_in[14];
  const float* Wv    = (const float*)d_in[15];
  const float* Wo    = (const float*)d_in[16];
  const float* lnFg  = (const float*)d_in[17];
  const float* lnFb  = (const float*)d_in[18];
  const float* W1    = (const float*)d_in[19];
  const float* b1    = (const float*)d_in[20];
  const float* W2    = (const float*)d_in[21];
  const float* b2    = (const float*)d_in[22];
  float* out = (float*)d_out;

  float* ws = (float*)d_ws;
  const size_t SL = (size_t)Md * Dd;   // 4M floats per slab
  float* tmp = ws;                     // slab 0: conv+ReLU out / lnF output
  float* hbf = ws + SL;                // slab 1: h
  float* inp = ws + 2 * SL;            // slab 2: inputs (residual)
  float* xn  = ws + 3 * SL;            // slab 3: xn, later ctx
  float* q   = ws + 4 * SL;            // slab 4: q, later attn out
  float* kk  = ws + 5 * SL;            // slab 5: k, later inter (bf16, slabs 5-6)
  float* vv  = ws + 6 * SL;            // slab 6: v
  bf16*  inter = (bf16*)kk;            // 4096x4096 bf16 = 32MB spans slabs 5-6

  const dim3 blk(128);
  const dim3 gD(Md / 64, Dd / 64);     // (64, 16)
  const dim3 gF(Md / 64, DFFd / 64);   // (64, 64)

  // conv1 -> relu (fused) -> LN1
  conv_gemm_kernel<<<gD, blk, 0, stream>>>(x, c1w, c1b, tmp);
  ln_kernel<<<Md, 256, 0, stream>>>(tmp, hbf, ln1g, ln1b);
  // conv2 -> relu -> LN2
  conv_gemm_kernel<<<gD, blk, 0, stream>>>(hbf, c2w, c2b, tmp);
  ln_kernel<<<Md, 256, 0, stream>>>(tmp, hbf, ln2g, ln2b);
  // inputs = h + x ; xn = LN_A(inputs)
  add_ln_kernel<<<Md, 256, 0, stream>>>(hbf, x, inp, xn, lnAg, lnAb);
  // Q (scaled 1/sqrt(DH)=1/8), K, V projections
  gemm_wmma_kernel<float, float><<<gD, blk, 0, stream>>>(
      xn, Wq, q, Dd, Dd, 0.125f, nullptr, 0, nullptr, nullptr);
  gemm_wmma_kernel<float, float><<<gD, blk, 0, stream>>>(
      xn, Wk, kk, Dd, Dd, 1.0f, nullptr, 0, nullptr, nullptr);
  gemm_wmma_kernel<float, float><<<gD, blk, 0, stream>>>(
      xn, Wv, vv, Dd, Dd, 1.0f, nullptr, 0, nullptr, nullptr);
  // attention -> ctx (into xn slab)
  attn_kernel<<<2 * Hd * (Sd / 16), 32, 0, stream>>>(q, kk, vv, amask, xn);
  // out = ctx @ Wo^T + inputs  (into q slab)
  gemm_wmma_kernel<float, float><<<gD, blk, 0, stream>>>(
      xn, Wo, q, Dd, Dd, 1.0f, nullptr, 0, nullptr, inp);
  // xf = LN_F(out) (into tmp slab)
  ln_kernel<<<Md, 256, 0, stream>>>(q, tmp, lnFg, lnFb);
  // inter = relu(xf @ W1^T + b1)  (bf16, slabs 5-6)
  gemm_wmma_kernel<float, bf16><<<gF, blk, 0, stream>>>(
      tmp, W1, inter, DFFd, Dd, 1.0f, b1, 1, nullptr, nullptr);
  // d_out = (inter @ W2^T + b2) * ffn_mask + out
  gemm_wmma_kernel<bf16, float><<<gD, blk, 0, stream>>>(
      inter, W2, out, Dd, DFFd, 1.0f, b2, 0, fmask, q);
}
